// GATv2Base_44744969289866
// MI455X (gfx1250) — matrix operand order repaired and checked by reading, not compiled, not run on hardware
//
#include <hip/hip_runtime.h>
#include <hip/hip_bf16.h>

// ---------------------------------------------------------------------------
// GATv2 (2 layers, H=4, C=64, HC=256) forward for MI455X / gfx1250.
// Dense GEMMs (h@Wl, h@Wr): v_wmma_f32_16x16x32_bf16 (bf16 in, f32 acc),
//   with the A strip staged into LDS via gfx1250 async copy
//   (global_load_async_to_lds_b128 + s_wait_asynccnt) and 2x A-frag reuse.
// Edge softmax/aggregation: wave-per-edge, 8-lane/head shuffle reductions,
// L2-resident float atomics (xl/xr/acc ~154MB < 192MB L2).
// ---------------------------------------------------------------------------

#define N_   50000
#define E_   800000
#define EP_  (E_ + N_)     // edges + self loops = 850000
#define NG_  32
#define HC_  256

typedef __attribute__((ext_vector_type(16))) __bf16 v16bf;
typedef __attribute__((ext_vector_type(8)))  float  v8f;

#if defined(__HIP_DEVICE_COMPILE__) && defined(__gfx1250__)
#define GFX1250_ASM 1
#else
#define GFX1250_ASM 0
#endif

static __device__ __forceinline__ unsigned short f32_to_bf16(float f) {
  unsigned u = __float_as_uint(f);
  u += 0x7FFFu + ((u >> 16) & 1u);       // round-to-nearest-even
  return (unsigned short)(u >> 16);
}

static __device__ __forceinline__ void atomicMaxFloat(float* addr, float value) {
  if (value >= 0.0f)
    atomicMax((int*)addr, __float_as_int(value));
  else
    atomicMin((unsigned int*)addr, __float_as_uint(value));
}

// ---------------------------------------------------------------------------
// Fill buffer with constant (used for -inf init of segment-max)
__global__ void fill_f32(float* __restrict__ p, float v, int n) {
  int t = blockIdx.x * blockDim.x + threadIdx.x;
  if (t < n) p[t] = v;
}

// ---------------------------------------------------------------------------
// Degree + self-loop attr:  deg[d] += 1, loopsum[d] += eattr[e]
__global__ void deg_loop_accum(const int* __restrict__ dst, const float* __restrict__ eattr,
                               float* __restrict__ deg, float* __restrict__ loopsum, int E) {
  int t = blockIdx.x * blockDim.x + threadIdx.x;
  if (t >= E) return;
  int d = dst[t];
  atomicAdd(&deg[d], 1.0f);
  atomicAdd(&loopsum[d], eattr[t]);
}
__global__ void loop_div(float* __restrict__ loopattr, const float* __restrict__ deg, int n) {
  int t = blockIdx.x * blockDim.x + threadIdx.x;
  if (t < n) loopattr[t] = loopattr[t] / fmaxf(deg[t], 1.0f);
}

// ---------------------------------------------------------------------------
// Encoder: h = relu(x @ enc_w + enc_b), x:[N,4] w:[4,64] -> bf16 [N,64]
__global__ void encoder_kernel(const float* __restrict__ x, const float* __restrict__ w,
                               const float* __restrict__ b, unsigned short* __restrict__ hb) {
  int t = blockIdx.x * blockDim.x + threadIdx.x;
  if (t >= N_ * 64) return;
  int n = t >> 6, c = t & 63;
  const float* xp = x + (size_t)n * 4;
  float v = b[c] + xp[0] * w[c] + xp[1] * w[64 + c] + xp[2] * w[128 + c] + xp[3] * w[192 + c];
  hb[t] = f32_to_bf16(fmaxf(v, 0.0f));
}

// ---------------------------------------------------------------------------
// Pack weight [K,256] f32 (row major) into per-lane WMMA B fragments (bf16).
// Layout: ((nt*nk + kt)*32 + lane)*16 + j ; within tile: k = 2*(j>>1)+(j&1)+16*(lane>>4)
__global__ void pack_b_kernel(const float* __restrict__ W, unsigned short* __restrict__ out, int K) {
  int t = blockIdx.x * blockDim.x + threadIdx.x;
  int total = K * 256;
  if (t >= total) return;
  int j     = t & 15;
  int lane  = (t >> 4) & 31;
  int ktnt  = t >> 9;
  int nk    = K >> 5;
  int kt    = ktnt % nk;
  int nt    = ktnt / nk;
  int khalf = lane >> 4;
  int n = nt * 16 + (lane & 15);
  int k = kt * 32 + khalf * 16 + ((j >> 1) << 1) + (j & 1);
  out[t] = f32_to_bf16(W[(size_t)k * 256 + n]);
}

// ---------------------------------------------------------------------------
// C[M,256] = A[M,K](bf16) @ W[K,256](packed bf16) + bias   via WMMA bf16.
// One block (8 waves) per 16-row m-tile; block covers all 16 n-tiles
// (2 per wave). A strip [16 x K] staged into LDS with async copy; LDS row
// pitch padded to K*2+16 bytes so the 16-row fragment reads are conflict-free.
__global__ __launch_bounds__(256)
void wmma_gemm_kernel(const unsigned short* __restrict__ A,
                      const unsigned short* __restrict__ Bp,
                      const float* __restrict__ bias,
                      float* __restrict__ C, int M, int K) {
  __shared__ __align__(16) unsigned char Alds[16 * (512 + 16)];   // max K=256
  const int tid   = threadIdx.x;
  const int wave  = tid >> 5;
  const int lane  = tid & 31;
  const int mtile = blockIdx.x;
  const int pitch = K * 2 + 16;          // bytes; (pitch/4)%64 != 0 -> no conflicts
  const int rowBytes = K * 2;            // 128 (K=64) or 512 (K=256)
  const int chunks = 16 * (rowBytes >> 4);

  // ---- stage A strip [16 x K] into LDS, 16B per lane, coalesced ----
  for (int c = tid; c < chunks; c += 256) {
    int byteoff = c << 4;
    int row = byteoff / rowBytes;
    int col = byteoff - row * rowBytes;
    const unsigned short* gsrc = A + (size_t)(mtile * 16 + row) * K + (col >> 1);
    unsigned ldsoff = (unsigned)(row * pitch + col);
#if GFX1250_ASM
    unsigned lds_v = (unsigned)(unsigned long long)(&Alds[0]) + ldsoff;
    asm volatile("global_load_async_to_lds_b128 %0, %1, off"
                 :: "v"(lds_v), "v"(gsrc) : "memory");
#else
    uint4 d = *(const uint4*)gsrc;
    *(uint4*)&Alds[ldsoff] = d;
#endif
  }
#if GFX1250_ASM
  asm volatile("s_wait_asynccnt 0x0" ::: "memory");   // own wave's async copies done
#endif
  __syncthreads();                                    // publish to all waves

  const int m     = lane & 15;
  const int khalf = lane >> 4;
  const int nk    = K >> 5;
  const int nt0   = wave * 2;            // this wave's two n-tiles

  const unsigned short* B0 = Bp + ((size_t)(nt0 * nk) * 32 + lane) * 16;
  const unsigned short* B1 = Bp + ((size_t)((nt0 + 1) * nk) * 32 + lane) * 16;

  v8f acc0 = {}, acc1 = {};
  for (int kt = 0; kt < nk; ++kt) {
    // A fragment from LDS: contiguous k = 16*khalf + s, matching packed B (8 ds dword loads)
    union { unsigned u[8]; v16bf v; } a;
    int kb = kt * 32 + 16 * khalf;
#pragma unroll
    for (int i = 0; i < 8; ++i) {
      int kloc = kb + (i << 1);
      a.u[i] = *(const unsigned*)&Alds[m * pitch + kloc * 2];
    }
    // B fragments: pre-packed, contiguous 32B per lane
    union { unsigned u[8]; v16bf v; } b0, b1;
    const unsigned* p0 = (const unsigned*)(B0 + (size_t)kt * 512);
    const unsigned* p1 = (const unsigned*)(B1 + (size_t)kt * 512);
#pragma unroll
    for (int i = 0; i < 8; ++i) { b0.u[i] = p0[i]; b1.u[i] = p1[i]; }

    acc0 = __builtin_amdgcn_wmma_f32_16x16x32_bf16(false, a.v, false, b0.v,
                                                   (short)0, acc0, false, false);
    acc1 = __builtin_amdgcn_wmma_f32_16x16x32_bf16(false, a.v, false, b1.v,
                                                   (short)0, acc1, false, false);
  }

#pragma unroll
  for (int tsel = 0; tsel < 2; ++tsel) {
    v8f acc = tsel ? acc1 : acc0;
    int n = (nt0 + tsel) * 16 + (lane & 15);
    float bn = bias[n];
#pragma unroll
    for (int r = 0; r < 8; ++r) {
      int mm = mtile * 16 + r + 8 * khalf;   // D layout: VGPR r -> row r (+8 upper half)
      C[(size_t)mm * 256 + n] = acc[r] + bn;
    }
  }
}

// ---------------------------------------------------------------------------
// Edge pass 1: alpha[e,h] = att . leaky_relu(xl[src]+xr[dst]+ea*we) ; segment max
__global__ __launch_bounds__(256)
void edge_pass1(const float* __restrict__ xl, const float* __restrict__ xr,
                const int* __restrict__ src0, const int* __restrict__ dst0,
                const float* __restrict__ eattr, const float* __restrict__ loopattr,
                const float* __restrict__ we, const float* __restrict__ att,
                float* __restrict__ alpha, float* __restrict__ amax) {
  int wid  = (blockIdx.x * blockDim.x + threadIdx.x) >> 5;
  int lane = threadIdx.x & 31;
  if (wid >= EP_) return;
  int s, d; float ea;
  if (wid < E_) { s = src0[wid]; d = dst0[wid]; ea = eattr[wid]; }
  else          { s = wid - E_;  d = s;         ea = loopattr[s]; }

  int c0 = lane * 8;
  const float4* xls = (const float4*)(xl + (size_t)s * 256 + c0);
  const float4* xrd = (const float4*)(xr + (size_t)d * 256 + c0);
  const float4* wev = (const float4*)(we + c0);
  const float4* atv = (const float4*)(att + c0);

  float partial = 0.0f;
#pragma unroll
  for (int half = 0; half < 2; ++half) {
    float4 a4 = xls[half], b4 = xrd[half], w4 = wev[half], t4 = atv[half];
    float v;
    v = a4.x + b4.x + ea * w4.x; v = v > 0.f ? v : 0.2f * v; partial += v * t4.x;
    v = a4.y + b4.y + ea * w4.y; v = v > 0.f ? v : 0.2f * v; partial += v * t4.y;
    v = a4.z + b4.z + ea * w4.z; v = v > 0.f ? v : 0.2f * v; partial += v * t4.z;
    v = a4.w + b4.w + ea * w4.w; v = v > 0.f ? v : 0.2f * v; partial += v * t4.w;
  }
  // reduce within 8-lane head group (head = lane>>3)
  partial += __shfl_xor(partial, 1, 32);
  partial += __shfl_xor(partial, 2, 32);
  partial += __shfl_xor(partial, 4, 32);
  if ((lane & 7) == 0) {
    int h = lane >> 3;
    alpha[(size_t)wid * 4 + h] = partial;
    atomicMaxFloat(&amax[(size_t)d * 4 + h], partial);
  }
}

// Edge pass 2: ex = exp(alpha - amax[dst]); den[dst] += ex
__global__ void edge_pass2(const int* __restrict__ dst0, const float* __restrict__ amax,
                           float* __restrict__ alpha, float* __restrict__ den) {
  int t = blockIdx.x * blockDim.x + threadIdx.x;
  if (t >= EP_ * 4) return;
  int e = t >> 2, h = t & 3;
  int d = (e < E_) ? dst0[e] : (e - E_);
  float ex = expf(alpha[t] - amax[(size_t)d * 4 + h]);
  alpha[t] = ex;
  atomicAdd(&den[(size_t)d * 4 + h], ex);
}

// Edge pass 3: acc[dst] += (ex/den[dst]) * xl[src]
__global__ __launch_bounds__(256)
void edge_pass3(const float* __restrict__ xl,
                const int* __restrict__ src0, const int* __restrict__ dst0,
                const float* __restrict__ alpha, const float* __restrict__ den,
                float* __restrict__ acc) {
  int wid  = (blockIdx.x * blockDim.x + threadIdx.x) >> 5;
  int lane = threadIdx.x & 31;
  if (wid >= EP_) return;
  int s, d;
  if (wid < E_) { s = src0[wid]; d = dst0[wid]; }
  else          { s = wid - E_;  d = s; }
  int h = lane >> 3;
  float a = alpha[(size_t)wid * 4 + h] / den[(size_t)d * 4 + h];
  int c0 = lane * 8;
  const float* xs = xl + (size_t)s * 256 + c0;
  float* ad = acc + (size_t)d * 256 + c0;
#pragma unroll
  for (int j = 0; j < 8; ++j) atomicAdd(&ad[j], a * xs[j]);
}

// ---------------------------------------------------------------------------
// Layer-1 epilogue: relu(acc + bias) -> bf16 input of layer 2
__global__ void bias_relu_bf16(const float* __restrict__ acc, const float* __restrict__ bias,
                               unsigned short* __restrict__ hb) {
  int t = blockIdx.x * blockDim.x + threadIdx.x;
  if (t >= N_ * 256) return;
  float v = acc[t] + bias[t & 255];
  hb[t] = f32_to_bf16(fmaxf(v, 0.0f));
}

// Layer-2 epilogue fused with pooling: pooled[batch[n]] += relu(acc + bias)
__global__ void bias_relu_pool(const float* __restrict__ acc, const float* __restrict__ bias,
                               const int* __restrict__ batch, float* __restrict__ pooled) {
  int t = blockIdx.x * blockDim.x + threadIdx.x;
  if (t >= N_ * 256) return;
  int n = t >> 8, c = t & 255;
  float v = fmaxf(acc[t] + bias[c], 0.0f);
  atomicAdd(&pooled[(size_t)batch[n] * 256 + c], v);
}

__global__ void count_nodes(const int* __restrict__ batch, float* __restrict__ counts) {
  int t = blockIdx.x * blockDim.x + threadIdx.x;
  if (t < N_) atomicAdd(&counts[batch[t]], 1.0f);
}
__global__ void pool_div(float* __restrict__ pooled, const float* __restrict__ counts) {
  int t = blockIdx.x * blockDim.x + threadIdx.x;
  if (t >= NG_ * 256) return;
  pooled[t] /= fmaxf(counts[t >> 8], 1.0f);
}

// ---------------------------------------------------------------------------
// Post MLP: z = relu(LN(g@p1_w + b)) then relu(z@p2_w + b)  (32 rows)
__global__ void mlp1_kernel(const float* __restrict__ g, const float* __restrict__ w,
                            const float* __restrict__ b, const float* __restrict__ lng,
                            const float* __restrict__ lnb, float* __restrict__ z) {
  __shared__ float gs[256];
  __shared__ float red[128];
  int row = blockIdx.x, c = threadIdx.x;
  gs[c]       = g[(size_t)row * 256 + c];
  gs[c + 128] = g[(size_t)row * 256 + 128 + c];
  __syncthreads();
  float acc = b[c];
  for (int k = 0; k < 256; ++k) acc += gs[k] * w[(size_t)k * 128 + c];
  red[c] = acc; __syncthreads();
  for (int s = 64; s > 0; s >>= 1) { if (c < s) red[c] += red[c + s]; __syncthreads(); }
  float mu = red[0] / 128.0f; __syncthreads();
  float dc = acc - mu;
  red[c] = dc * dc; __syncthreads();
  for (int s = 64; s > 0; s >>= 1) { if (c < s) red[c] += red[c + s]; __syncthreads(); }
  float var = red[0] / 128.0f;
  float zn = dc * rsqrtf(var + 1e-5f) * lng[c] + lnb[c];
  z[(size_t)row * 128 + c] = fmaxf(zn, 0.0f);
}

__global__ void mlp2_kernel(const float* __restrict__ z, const float* __restrict__ w,
                            const float* __restrict__ b, float* __restrict__ out) {
  __shared__ float zs[128];
  int row = blockIdx.x, c = threadIdx.x;   // 64 threads
  zs[c]      = z[(size_t)row * 128 + c];
  zs[c + 64] = z[(size_t)row * 128 + 64 + c];
  __syncthreads();
  float acc = b[c];
  for (int k = 0; k < 128; ++k) acc += zs[k] * w[(size_t)k * 64 + c];
  out[(size_t)row * 64 + c] = fmaxf(acc, 0.0f);
}

// ---------------------------------------------------------------------------
static inline int cdiv(long long a, long long b) { return (int)((a + b - 1) / b); }

extern "C" void kernel_launch(void* const* d_in, const int* in_sizes, int n_in,
                              void* d_out, int out_size, void* d_ws, size_t ws_size,
                              hipStream_t stream) {
  const float* x       = (const float*)d_in[0];
  const int*   eidx    = (const int*)  d_in[1];
  const float* eattr   = (const float*)d_in[2];
  const int*   batch   = (const int*)  d_in[3];
  const float* enc_w   = (const float*)d_in[4];
  const float* enc_b   = (const float*)d_in[5];
  const float* g1_wl   = (const float*)d_in[6];
  const float* g1_bl   = (const float*)d_in[7];
  const float* g1_wr   = (const float*)d_in[8];
  const float* g1_br   = (const float*)d_in[9];
  const float* g1_we   = (const float*)d_in[10];
  const float* g1_att  = (const float*)d_in[11];
  const float* g1_bias = (const float*)d_in[12];
  const float* g2_wl   = (const float*)d_in[13];
  const float* g2_bl   = (const float*)d_in[14];
  const float* g2_wr   = (const float*)d_in[15];
  const float* g2_br   = (const float*)d_in[16];
  const float* g2_we   = (const float*)d_in[17];
  const float* g2_att  = (const float*)d_in[18];
  const float* g2_bias = (const float*)d_in[19];
  const float* p1_w    = (const float*)d_in[20];
  const float* p1_b    = (const float*)d_in[21];
  const float* ln_g    = (const float*)d_in[22];
  const float* ln_b    = (const float*)d_in[23];
  const float* p2_w    = (const float*)d_in[24];
  const float* p2_b    = (const float*)d_in[25];

  const int* srcp = eidx;
  const int* dstp = eidx + E_;

  // ---- workspace layout ----
  char* w = (char*)d_ws;
  size_t off = 0;
  auto alloc = [&](size_t bytes) -> void* {
    void* p = w + off;
    off = (off + bytes + 255) & ~(size_t)255;
    return p;
  };
  float* deg      = (float*)alloc((size_t)N_ * 4);
  float* loopattr = (float*)alloc((size_t)N_ * 4);
  unsigned short* hb1  = (unsigned short*)alloc((size_t)N_ * 64 * 2);
  unsigned short* hb2  = (unsigned short*)alloc((size_t)N_ * 256 * 2);
  unsigned short* pk1l = (unsigned short*)alloc((size_t)64 * 256 * 2);
  unsigned short* pk1r = (unsigned short*)alloc((size_t)64 * 256 * 2);
  unsigned short* pk2l = (unsigned short*)alloc((size_t)256 * 256 * 2);
  unsigned short* pk2r = (unsigned short*)alloc((size_t)256 * 256 * 2);
  float* xl    = (float*)alloc((size_t)N_ * 256 * 4);
  float* xr    = (float*)alloc((size_t)N_ * 256 * 4);
  float* acc   = (float*)alloc((size_t)N_ * 256 * 4);
  float* amax  = (float*)alloc((size_t)N_ * 4 * 4);
  float* den   = (float*)alloc((size_t)N_ * 4 * 4);
  float* alpha = (float*)alloc((size_t)EP_ * 4 * 4);
  float* pooled= (float*)alloc((size_t)NG_ * 256 * 4);
  float* counts= (float*)alloc((size_t)NG_ * 4);
  float* zbuf  = (float*)alloc((size_t)NG_ * 128 * 4);

  const int B = 256;
  const int gemm_blocks = N_ / 16;   // one block per 16-row m-tile (3125)

  // 0) self-loop attributes
  hipMemsetAsync(deg, 0, (size_t)N_ * 4, stream);
  hipMemsetAsync(loopattr, 0, (size_t)N_ * 4, stream);
  deg_loop_accum<<<cdiv(E_, B), B, 0, stream>>>(dstp, eattr, deg, loopattr, E_);
  loop_div<<<cdiv(N_, B), B, 0, stream>>>(loopattr, deg, N_);

  // 1) encoder -> bf16
  encoder_kernel<<<cdiv((long long)N_ * 64, B), B, 0, stream>>>(x, enc_w, enc_b, hb1);

  // 2) pack weights to WMMA B-fragment layout
  pack_b_kernel<<<cdiv(64 * 256, B), B, 0, stream>>>(g1_wl, pk1l, 64);
  pack_b_kernel<<<cdiv(64 * 256, B), B, 0, stream>>>(g1_wr, pk1r, 64);
  pack_b_kernel<<<cdiv(256 * 256, B), B, 0, stream>>>(g2_wl, pk2l, 256);
  pack_b_kernel<<<cdiv(256 * 256, B), B, 0, stream>>>(g2_wr, pk2r, 256);

  // ---- GATv2 layer 1 ----
  wmma_gemm_kernel<<<gemm_blocks, B, 0, stream>>>(hb1, pk1l, g1_bl, xl, N_, 64);
  wmma_gemm_kernel<<<gemm_blocks, B, 0, stream>>>(hb1, pk1r, g1_br, xr, N_, 64);

  fill_f32<<<cdiv(N_ * 4, B), B, 0, stream>>>(amax, -__builtin_huge_valf(), N_ * 4);
  hipMemsetAsync(den, 0, (size_t)N_ * 16, stream);
  edge_pass1<<<cdiv(EP_, 8), B, 0, stream>>>(xl, xr, srcp, dstp, eattr, loopattr,
                                             g1_we, g1_att, alpha, amax);
  edge_pass2<<<cdiv((long long)EP_ * 4, B), B, 0, stream>>>(dstp, amax, alpha, den);
  hipMemsetAsync(acc, 0, (size_t)N_ * 1024, stream);
  edge_pass3<<<cdiv(EP_, 8), B, 0, stream>>>(xl, srcp, dstp, alpha, den, acc);
  bias_relu_bf16<<<cdiv((long long)N_ * 256, B), B, 0, stream>>>(acc, g1_bias, hb2);

  // ---- GATv2 layer 2 ----
  wmma_gemm_kernel<<<gemm_blocks, B, 0, stream>>>(hb2, pk2l, g2_bl, xl, N_, 256);
  wmma_gemm_kernel<<<gemm_blocks, B, 0, stream>>>(hb2, pk2r, g2_br, xr, N_, 256);

  fill_f32<<<cdiv(N_ * 4, B), B, 0, stream>>>(amax, -__builtin_huge_valf(), N_ * 4);
  hipMemsetAsync(den, 0, (size_t)N_ * 16, stream);
  edge_pass1<<<cdiv(EP_, 8), B, 0, stream>>>(xl, xr, srcp, dstp, eattr, loopattr,
                                             g2_we, g2_att, alpha, amax);
  edge_pass2<<<cdiv((long long)EP_ * 4, B), B, 0, stream>>>(dstp, amax, alpha, den);
  hipMemsetAsync(acc, 0, (size_t)N_ * 1024, stream);
  edge_pass3<<<cdiv(EP_, 8), B, 0, stream>>>(xl, srcp, dstp, alpha, den, acc);

  // ---- pooling (fused with layer-2 bias+relu) ----
  hipMemsetAsync(pooled, 0, (size_t)NG_ * 1024, stream);
  hipMemsetAsync(counts, 0, (size_t)NG_ * 4, stream);
  bias_relu_pool<<<cdiv((long long)N_ * 256, B), B, 0, stream>>>(acc, g2_bias, batch, pooled);
  count_nodes<<<cdiv(N_, B), B, 0, stream>>>(batch, counts);
  pool_div<<<cdiv(NG_ * 256, B), B, 0, stream>>>(pooled, counts);

  // ---- post MLP ----
  mlp1_kernel<<<NG_, 128, 0, stream>>>(pooled, p1_w, p1_b, ln_g, ln_b, zbuf);
  mlp2_kernel<<<NG_, 64, 0, stream>>>(zbuf, p2_w, p2_b, (float*)d_out);
}
